// PolygonSegmenterGCNConv_58935541236088
// MI455X (gfx1250) — compile-verified, matrix operand order checked by reading, not run on hardware
//
#include <hip/hip_runtime.h>
#include <hip/hip_bf16.h>

#define NNODES 50000
#define NEDGES 800000
#define NLABEL 262144
#define BN_EPS 1e-5f

typedef __attribute__((ext_vector_type(16))) _Float16 v16h;
typedef __attribute__((ext_vector_type(8)))  float    v8f;

__device__ __forceinline__ void atomAddF(float* p, float v) {
    __hip_atomic_fetch_add(p, v, __ATOMIC_RELAXED, __HIP_MEMORY_SCOPE_AGENT);
}

// ---------------- WMMA fragment helpers (CDNA5 16x16x32 f16, f32 acc) ----------------
// A (16x32 f16): lanes 0-15 row M=lane K=0..7,16..23 ; lanes 16-31 row M=lane-16 K=8..15,24..31
__device__ __forceinline__ v16h load_a_f32(const float* A, int lda, int row0, int k0, int lane) {
    v16h a;
    int r  = row0 + (lane & 15);
    int kb = k0 + ((lane >> 4) << 3);
    const float* p = A + (size_t)r * lda + kb;
#pragma unroll
    for (int e = 0; e < 16; ++e) {
        int k = e + ((e >= 8) ? 8 : 0);
        a[e] = (_Float16)p[k];
    }
    return a;
}

// B (32x16 f16, row-major [K x N] in memory): lane holds column n = lane&15, same K striping as A
__device__ __forceinline__ v16h load_b_f16(const _Float16* B, int ldb, int k0, int col0, int lane) {
    v16h b;
    int c  = col0 + (lane & 15);
    int kb = k0 + ((lane >> 4) << 3);
    const _Float16* p = B + (size_t)kb * ldb + c;
#pragma unroll
    for (int e = 0; e < 16; ++e) {
        int k = e + ((e >= 8) ? 8 : 0);
        b[e] = p[(size_t)k * ldb];
    }
    return b;
}

// ---------------- small utility kernels ----------------
__global__ void k_f32_to_f16(const float* __restrict__ src, _Float16* __restrict__ dst, int n) {
    int i = blockIdx.x * blockDim.x + threadIdx.x;
    if (i < n) dst[i] = (_Float16)src[i];
}

__global__ void k_zero_f32(float* p, int n) {
    int i = blockIdx.x * blockDim.x + threadIdx.x;
    if (i < n) p[i] = 0.0f;
}

// out[i, f] = bias[f]   (F power of two)
__global__ void k_init_bias(float* __restrict__ out, const float* __restrict__ b, int total, int Fmask) {
    int i = blockIdx.x * blockDim.x + threadIdx.x;
    if (i < total) out[i] = b[i & Fmask];
}

// ---------------- degree / normalization ----------------
__global__ void k_degree(const int* __restrict__ col, const float* __restrict__ ew,
                         float* __restrict__ deg, int ne, int n) {
    int i = blockIdx.x * blockDim.x + threadIdx.x;
    int total = ne + n;
    if (i >= total) return;
    if (i < ne) atomAddF(&deg[col[i]], ew[i]);
    else        atomAddF(&deg[i - ne], 1.0f);   // self-loop weight 1
}

__global__ void k_dinv(const float* __restrict__ deg, float* __restrict__ dinv, int n) {
    int i = blockIdx.x * blockDim.x + threadIdx.x;
    if (i < n) {
        float d = deg[i];
        dinv[i] = (d > 0.0f) ? rsqrtf(d) : 0.0f;
    }
}

// ---------------- dense GEMM: C[MxN] = A[MxK](f32) * B[KxN](f16), f32 accum via WMMA ----------------
__global__ void k_gemm_wmma(const float* __restrict__ A, const _Float16* __restrict__ Bh,
                            float* __restrict__ C, int M, int N, int K) {
    int lane = threadIdx.x & 31;
    int wave = threadIdx.x >> 5;
    int tiles_n = N >> 4;
    int tile = blockIdx.x * (blockDim.x >> 5) + wave;
    int total = (M >> 4) * tiles_n;
    if (tile >= total) return;
    int tm = (tile / tiles_n) << 4;
    int tn = (tile % tiles_n) << 4;

    v8f acc = {};
    for (int k0 = 0; k0 < K; k0 += 32) {
        v16h a = load_a_f32(A, K, tm, k0, lane);
        v16h b = load_b_f16(Bh, N, k0, tn, lane);
        acc = __builtin_amdgcn_wmma_f32_16x16x32_f16(false, a, false, b, (short)0, acc, false, false);
    }
    int r0 = tm + ((lane >> 4) << 3);
    int cc = tn + (lane & 15);
#pragma unroll
    for (int v = 0; v < 8; ++v)
        C[(size_t)(r0 + v) * N + cc] = acc[v];
}

// ---------------- edge scatter: out[col] += xw[row] * dinv[row]*ew*dinv[col] ----------------
template <int F>
__global__ void k_scatter(const float* __restrict__ xw, const int* __restrict__ row,
                          const int* __restrict__ col, const float* __restrict__ ew,
                          const float* __restrict__ dinv, float* __restrict__ out,
                          int ne, int n) {
    constexpr int V = F / 32;
    int lane = threadIdx.x & 31;
    int wglob = (blockIdx.x * blockDim.x + threadIdx.x) >> 5;
    int nwaves = (gridDim.x * blockDim.x) >> 5;
    int total = ne + n;
    for (int e = wglob; e < total; e += nwaves) {
        int r, c; float w;
        if (e < ne) { r = row[e]; c = col[e]; w = ew[e]; }
        else        { r = e - ne; c = r;      w = 1.0f;  }
        float nrm = dinv[r] * w * dinv[c];
        const float* src = xw + (size_t)r * F + lane * V;
        float*       dst = out + (size_t)c * F + lane * V;
#pragma unroll
        for (int i = 0; i < V; ++i) atomAddF(dst + i, src[i] * nrm);
    }
}

// ---------------- BatchNorm (train) + ReLU, C = 128 channels ----------------
__global__ void k_bn_stats(const float* __restrict__ x, float* __restrict__ stats, int n) {
    int c = threadIdx.x;           // blockDim.x == 128
    float s = 0.0f, ss = 0.0f;
    for (int i = blockIdx.x; i < n; i += gridDim.x) {
        float v = x[(size_t)i * 128 + c];
        s += v; ss += v * v;
    }
    atomAddF(&stats[c], s);
    atomAddF(&stats[128 + c], ss);
}

__global__ void k_bn_apply_relu(float* __restrict__ x, const float* __restrict__ stats,
                                const float* __restrict__ gamma, const float* __restrict__ beta,
                                int n) {
    int i = blockIdx.x * blockDim.x + threadIdx.x;
    int total = n * 128;
    if (i >= total) return;
    int c = i & 127;
    float inv_n = 1.0f / (float)n;
    float m = stats[c] * inv_n;
    float var = stats[128 + c] * inv_n - m * m;
    float sc = gamma[c] * rsqrtf(var + BN_EPS);
    float v = (x[i] - m) * sc + beta[c];
    x[i] = fmaxf(v, 0.0f);
}

// ---------------- fused decoder: 16 label-edges per wave, two WMMA layers + dot ----------------
__global__ void k_decoder(const float* __restrict__ enc, const int* __restrict__ eli,
                          const _Float16* __restrict__ dw1h, const float* __restrict__ db1,
                          const _Float16* __restrict__ dw2h, const float* __restrict__ db2,
                          const float* __restrict__ dw3, const float* __restrict__ db3,
                          float* __restrict__ out, int L) {
    __shared__ _Float16 zbuf[8 * 16 * 64];   // one 16x64 f16 tile per wave
    int lane = threadIdx.x & 31;
    int wave = threadIdx.x >> 5;
    int tile = blockIdx.x * (blockDim.x >> 5) + wave;
    if (tile >= (L >> 4)) return;            // grid sized exactly; never taken
    _Float16* zb = zbuf + wave * (16 * 64);

    int m    = lane & 15;
    int hi   = lane >> 4;
    int srcn = eli[tile * 16 + m];
    int dstn = eli[L + tile * 16 + m];

    // ---- layer 1: z1 = relu([enc_src | enc_dst] @ dw1 + db1)  [16 x 64] ----
    v8f acc1[4] = {};
    for (int k0 = 0; k0 < 128; k0 += 32) {
        v16h a;
        int kb = k0 + (hi << 3);
#pragma unroll
        for (int e = 0; e < 16; ++e) {
            int k = kb + e + ((e >= 8) ? 8 : 0);
            float v = (k < 64) ? enc[(size_t)srcn * 64 + k]
                               : enc[(size_t)dstn * 64 + (k - 64)];
            a[e] = (_Float16)v;
        }
#pragma unroll
        for (int nt = 0; nt < 4; ++nt) {
            v16h b = load_b_f16(dw1h, 64, k0, nt * 16, lane);
            acc1[nt] = __builtin_amdgcn_wmma_f32_16x16x32_f16(false, a, false, b, (short)0, acc1[nt], false, false);
        }
    }
    // bias + relu -> LDS as f16 (re-layout C-frag -> A-frag source)
#pragma unroll
    for (int nt = 0; nt < 4; ++nt) {
        int ccol = nt * 16 + m;
        float bv = db1[ccol];
#pragma unroll
        for (int v = 0; v < 8; ++v) {
            int rr = v + 8 * hi;
            zb[rr * 64 + ccol] = (_Float16)fmaxf(acc1[nt][v] + bv, 0.0f);
        }
    }
    __syncthreads();

    // ---- layer 2: z2 = relu(z1 @ dw2 + db2)  [16 x 64] ----
    v8f acc2[4] = {};
    for (int k0 = 0; k0 < 64; k0 += 32) {
        v16h a;
        int kb = k0 + (hi << 3);
#pragma unroll
        for (int e = 0; e < 16; ++e) {
            int k = kb + e + ((e >= 8) ? 8 : 0);
            a[e] = zb[m * 64 + k];
        }
#pragma unroll
        for (int nt = 0; nt < 4; ++nt) {
            v16h b = load_b_f16(dw2h, 64, k0, nt * 16, lane);
            acc2[nt] = __builtin_amdgcn_wmma_f32_16x16x32_f16(false, a, false, b, (short)0, acc2[nt], false, false);
        }
    }

    // ---- layer 3: logit = relu(z2) @ dw3 + db3 ; half-wave reduction over 16 cols ----
    float acc3[8] = {};
#pragma unroll
    for (int nt = 0; nt < 4; ++nt) {
        int ccol = nt * 16 + m;
        float bv = db2[ccol];
        float wv = dw3[ccol];
#pragma unroll
        for (int v = 0; v < 8; ++v)
            acc3[v] += fmaxf(acc2[nt][v] + bv, 0.0f) * wv;
    }
#pragma unroll
    for (int msk = 1; msk < 16; msk <<= 1) {
#pragma unroll
        for (int v = 0; v < 8; ++v)
            acc3[v] += __shfl_xor(acc3[v], msk, 32);
    }
    if (m == 0) {
        float b3 = db3[0];
#pragma unroll
        for (int v = 0; v < 8; ++v)
            out[tile * 16 + v + 8 * hi] = acc3[v] + b3;
    }
}

// ---------------- host-side launch ----------------
static inline size_t align256(size_t x) { return (x + 255) & ~(size_t)255; }

extern "C" void kernel_launch(void* const* d_in, const int* in_sizes, int n_in,
                              void* d_out, int out_size, void* d_ws, size_t ws_size,
                              hipStream_t stream) {
    (void)in_sizes; (void)n_in; (void)out_size; (void)ws_size;

    const float* x   = (const float*)d_in[0];
    const int*   ei  = (const int*)d_in[1];        // [2, NEDGES]
    const float* ew  = (const float*)d_in[2];
    const int*   eli = (const int*)d_in[3];        // [2, NLABEL]
    const float* w1  = (const float*)d_in[4];  const float* b1 = (const float*)d_in[5];
    const float* w2  = (const float*)d_in[6];  const float* b2 = (const float*)d_in[7];
    const float* w3  = (const float*)d_in[8];  const float* b3 = (const float*)d_in[9];
    const float* w4  = (const float*)d_in[10]; const float* b4 = (const float*)d_in[11];
    const float* bn_gamma = (const float*)d_in[12];
    const float* bn_beta  = (const float*)d_in[13];
    const float* dw1 = (const float*)d_in[14]; const float* db1 = (const float*)d_in[15];
    const float* dw2 = (const float*)d_in[16]; const float* db2 = (const float*)d_in[17];
    const float* dw3 = (const float*)d_in[18]; const float* db3 = (const float*)d_in[19];
    float* out = (float*)d_out;

    const int* erow = ei;
    const int* ecol = ei + NEDGES;

    // workspace carve-up
    char* ws = (char*)d_ws;
    float* bufA = (float*)ws; ws += align256((size_t)NNODES * 128 * 4);
    float* bufB = (float*)ws; ws += align256((size_t)NNODES * 128 * 4);
    float* bufC = (float*)ws; ws += align256((size_t)NNODES * 128 * 4);
    float* deg  = (float*)ws; ws += align256((size_t)NNODES * 4);
    float* dinv = (float*)ws; ws += align256((size_t)NNODES * 4);
    float* stats = (float*)ws; ws += align256(256 * 4);
    _Float16* w1h  = (_Float16*)ws; ws += align256(32 * 128 * 2);
    _Float16* w2h  = (_Float16*)ws; ws += align256(128 * 128 * 2);
    _Float16* w3h  = (_Float16*)ws; ws += align256(128 * 128 * 2);
    _Float16* w4h  = (_Float16*)ws; ws += align256(128 * 64 * 2);
    _Float16* dw1h = (_Float16*)ws; ws += align256(128 * 64 * 2);
    _Float16* dw2h = (_Float16*)ws; ws += align256(64 * 64 * 2);

    const int T = 256;
    auto cdiv = [](int a, int b) { return (a + b - 1) / b; };

    // weights -> f16
    k_f32_to_f16<<<cdiv(32 * 128, T), T, 0, stream>>>(w1, w1h, 32 * 128);
    k_f32_to_f16<<<cdiv(128 * 128, T), T, 0, stream>>>(w2, w2h, 128 * 128);
    k_f32_to_f16<<<cdiv(128 * 128, T), T, 0, stream>>>(w3, w3h, 128 * 128);
    k_f32_to_f16<<<cdiv(128 * 64, T), T, 0, stream>>>(w4, w4h, 128 * 64);
    k_f32_to_f16<<<cdiv(128 * 64, T), T, 0, stream>>>(dw1, dw1h, 128 * 64);
    k_f32_to_f16<<<cdiv(64 * 64, T), T, 0, stream>>>(dw2, dw2h, 64 * 64);

    // degrees / norm factors (same for all layers)
    k_zero_f32<<<cdiv(NNODES, T), T, 0, stream>>>(deg, NNODES);
    k_degree<<<cdiv(NEDGES + NNODES, T), T, 0, stream>>>(ecol, ew, deg, NEDGES, NNODES);
    k_dinv<<<cdiv(NNODES, T), T, 0, stream>>>(deg, dinv, NNODES);

    const int gemm_waves = T / 32;
    const int scat_blocks = 2048;

    // ---- layer 1: x[.,32] -> bufB[.,128] ----
    k_gemm_wmma<<<cdiv((NNODES / 16) * (128 / 16), gemm_waves), T, 0, stream>>>(x, w1h, bufA, NNODES, 128, 32);
    k_init_bias<<<cdiv(NNODES * 128, T), T, 0, stream>>>(bufB, b1, NNODES * 128, 127);
    k_scatter<128><<<scat_blocks, T, 0, stream>>>(bufA, erow, ecol, ew, dinv, bufB, NEDGES, NNODES);
    k_zero_f32<<<1, 256, 0, stream>>>(stats, 256);
    k_bn_stats<<<512, 128, 0, stream>>>(bufB, stats, NNODES);
    k_bn_apply_relu<<<cdiv(NNODES * 128, T), T, 0, stream>>>(bufB, stats, bn_gamma + 0, bn_beta + 0, NNODES);

    // ---- layer 2: bufB -> bufC ----
    k_gemm_wmma<<<cdiv((NNODES / 16) * (128 / 16), gemm_waves), T, 0, stream>>>(bufB, w2h, bufA, NNODES, 128, 128);
    k_init_bias<<<cdiv(NNODES * 128, T), T, 0, stream>>>(bufC, b2, NNODES * 128, 127);
    k_scatter<128><<<scat_blocks, T, 0, stream>>>(bufA, erow, ecol, ew, dinv, bufC, NEDGES, NNODES);
    k_zero_f32<<<1, 256, 0, stream>>>(stats, 256);
    k_bn_stats<<<512, 128, 0, stream>>>(bufC, stats, NNODES);
    k_bn_apply_relu<<<cdiv(NNODES * 128, T), T, 0, stream>>>(bufC, stats, bn_gamma + 128, bn_beta + 128, NNODES);

    // ---- layer 3: bufC -> bufB ----
    k_gemm_wmma<<<cdiv((NNODES / 16) * (128 / 16), gemm_waves), T, 0, stream>>>(bufC, w3h, bufA, NNODES, 128, 128);
    k_init_bias<<<cdiv(NNODES * 128, T), T, 0, stream>>>(bufB, b3, NNODES * 128, 127);
    k_scatter<128><<<scat_blocks, T, 0, stream>>>(bufA, erow, ecol, ew, dinv, bufB, NEDGES, NNODES);
    k_zero_f32<<<1, 256, 0, stream>>>(stats, 256);
    k_bn_stats<<<512, 128, 0, stream>>>(bufB, stats, NNODES);
    k_bn_apply_relu<<<cdiv(NNODES * 128, T), T, 0, stream>>>(bufB, stats, bn_gamma + 256, bn_beta + 256, NNODES);

    // ---- layer 4 (encoder output, 64-wide, no BN/ReLU): bufB -> bufC[.,64] ----
    k_gemm_wmma<<<cdiv((NNODES / 16) * (64 / 16), gemm_waves), T, 0, stream>>>(bufB, w4h, bufA, NNODES, 64, 128);
    k_init_bias<<<cdiv(NNODES * 64, T), T, 0, stream>>>(bufC, b4, NNODES * 64, 63);
    k_scatter<64><<<scat_blocks, T, 0, stream>>>(bufA, erow, ecol, ew, dinv, bufC, NEDGES, NNODES);

    // ---- fused decoder MLP over candidate edges ----
    k_decoder<<<(NLABEL / 16) / gemm_waves, T, 0, stream>>>(bufC, eli, dw1h, db1, dw2h, db2, dw3, db3, out, NLABEL);
}
